// MultiHeadAttention_20907900797029
// MI455X (gfx1250) — matrix operand order repaired and checked by reading, not compiled, not run on hardware
//
#include <hip/hip_runtime.h>
#include <math.h>

// MHA: B=2, S=2048, D=1024, H=16, DQ=DK=64.
// Pipeline (all f16 WMMA, f32 accumulate):
//   1) gemm (f32 A)  : Q/K/V projections  -> f16 [B*S, H*64] in d_ws
//   2) flash attention (online softmax, TDM-staged V tiles) -> f16 in d_ws
//   3) gemm (f16 A)  : output projection  -> f32 d_out
// Workspace use: 4 x 4096*1024 halves = 32 MB.

typedef __attribute__((ext_vector_type(16))) _Float16     v16h;
typedef __attribute__((ext_vector_type(8)))  _Float16     v8h;
typedef __attribute__((ext_vector_type(8)))  float        v8f;
typedef __attribute__((ext_vector_type(4)))  unsigned int u32x4;
typedef __attribute__((ext_vector_type(8)))  unsigned int u32x8;

#define SHUF16 0,1,2,3,4,5,6,7,8,9,10,11,12,13,14,15

__device__ __forceinline__ v8f wmma_f16(v16h a, v16h b, v8f c) {
  // D = A(16x32,f16) * B(32x16,f16) + C(16x16,f32)
  return __builtin_amdgcn_wmma_f32_16x16x32_f16(false, a, false, b, (short)0, c,
                                                false, false);
}

// ---- DPP 16-lane butterfly reductions (pure VALU, no ds_bpermute) ----------
template <int CTRL>
__device__ __forceinline__ float dpp_perm(float x) {
  return __int_as_float(__builtin_amdgcn_update_dpp(
      __float_as_int(x), __float_as_int(x), CTRL, 0xF, 0xF, true));
}
__device__ __forceinline__ float red_max16(float x) {
  x = fmaxf(x, dpp_perm<0xB1>(x));    // quad_perm(1,0,3,2)  : xor 1
  x = fmaxf(x, dpp_perm<0x4E>(x));    // quad_perm(2,3,0,1)  : xor 2
  x = fmaxf(x, dpp_perm<0x141>(x));   // row_half_mirror     : cross-quad in 8
  x = fmaxf(x, dpp_perm<0x140>(x));   // row_mirror          : cross-8 in 16
  return x;
}
__device__ __forceinline__ float red_sum16(float x) {
  x += dpp_perm<0xB1>(x);
  x += dpp_perm<0x4E>(x);
  x += dpp_perm<0x141>(x);
  x += dpp_perm<0x140>(x);
  return x;
}

// ---------------------------------------------------------------------------
// Generic tiled GEMM: C[M,N] = A[M,K] * W[K,N] + bias[N]
//   A: f32 (A_IS_F32) or f16, row-major.
//   W: always f32 in memory; staged to LDS as f16.
//     QKV_W: W stored [H, K, 64], logical W[k,n] = W[((n>>6)*K + k)*64 + (n&63)]
//     else : W row-major [K, N]
//   Out: f32 (OUT_F32) or f16, row-major [M, N].
// Block = 128 threads (4 waves) -> 64x64 output tile.
// ---------------------------------------------------------------------------
template<bool A_IS_F32, bool QKV_W, bool OUT_F32>
__global__ __launch_bounds__(128)
void gemm_kernel(const void* __restrict__ Aptr, const float* __restrict__ W,
                 const float* __restrict__ bias, void* __restrict__ Out) {
  constexpr int N = 1024, K = 1024;
  const int tid  = threadIdx.x;
  const int wave = tid >> 5;
  const int lane = tid & 31;
  const int hi   = lane >> 4;     // which 16-lane half
  const int l16  = lane & 15;
  const int kbA  = hi * 8;        // unused; A frag uses kbB too
  const int kbB  = hi * 16;       // B-frag K base
  const int col0 = blockIdx.x * 64;
  const int row0 = blockIdx.y * 64;
  const int aRow = row0 + wave * 16 + l16;

  __shared__ __align__(16) _Float16 ldsB[32 * 64];

  // Per-thread staging source (16 contiguous floats in either weight layout)
  const int stR = tid >> 2;
  const int stC = (tid & 3) * 16;

  v8f acc[4] = {};

  for (int k0 = 0; k0 < K; k0 += 32) {
    __syncthreads();
    {  // stage W tile [32 x 64] into LDS as f16 (vectorized)
      const float* wsrc;
      if (QKV_W) wsrc = W + ((size_t)(col0 >> 6) * K + (k0 + stR)) * 64 + stC;
      else       wsrc = W + (size_t)(k0 + stR) * N + col0 + stC;
      v8f w0 = *(const v8f*)(wsrc);
      v8f w1 = *(const v8f*)(wsrc + 8);
      v8h h0, h1;
#pragma unroll
      for (int i = 0; i < 8; ++i) { h0[i] = (_Float16)w0[i]; h1[i] = (_Float16)w1[i]; }
      *(v8h*)(&ldsB[stR * 64 + stC])     = h0;
      *(v8h*)(&ldsB[stR * 64 + stC + 8]) = h1;
      if (k0 + 32 < K) __builtin_prefetch(wsrc + (QKV_W ? 32 * 64 : 32 * N), 0, 3);
    }
    __syncthreads();

    // A fragment (16x32): elem j -> K = kbB + j
    v16h a;
    if constexpr (A_IS_F32) {
      const float* ab = (const float*)Aptr + (size_t)aRow * K + k0;
      v8f f0 = *(const v8f*)(ab + kbB);
      v8f f1 = *(const v8f*)(ab + kbB + 8);
#pragma unroll
      for (int i = 0; i < 8; ++i) { a[i] = (_Float16)f0[i]; a[8 + i] = (_Float16)f1[i]; }
      if (k0 + 32 < K) __builtin_prefetch(ab + 32, 0, 3);
    } else {
      const _Float16* ab = (const _Float16*)Aptr + (size_t)aRow * K + k0;
      v8h lo  = *(const v8h*)(ab + kbB);
      v8h hi8 = *(const v8h*)(ab + kbB + 8);
      a = __builtin_shufflevector(lo, hi8, SHUF16);
      if (k0 + 32 < K) __builtin_prefetch(ab + 32, 0, 3);
    }

#pragma unroll
    for (int et = 0; et < 4; ++et) {
      v16h bfrag;  // B frag: elem j -> K = kbB + j, col = et*16 + l16
#pragma unroll
      for (int j = 0; j < 16; ++j)
        bfrag[j] = ldsB[(kbB + j) * 64 + et * 16 + l16];
      acc[et] = wmma_f16(a, bfrag, acc[et]);
    }
  }

  // Epilogue: C layout -> VGPR r holds row (r + 8*hi), col = l16
#pragma unroll
  for (int et = 0; et < 4; ++et) {
#pragma unroll
    for (int r = 0; r < 8; ++r) {
      const int row = row0 + wave * 16 + r + hi * 8;
      const int col = col0 + et * 16 + l16;
      const float v = acc[et][r] + bias[col];
      if constexpr (OUT_F32) ((float*)Out)[(size_t)row * N + col] = v;
      else ((_Float16*)Out)[(size_t)row * N + col] = (_Float16)v;
    }
  }
}

// ---------------------------------------------------------------------------
// Flash attention over f16 Q/K/V stored [B*S, H*64].
// Grid: (S/64, H, B); block 128 = 4 waves, wave handles 16 query rows.
// V tiles staged to LDS by the Tensor Data Mover (tensor_load_to_lds),
// key loop step = 32: 4 WMMAs Q*K^T, DPP online softmax, P via per-wave LDS
// (C-layout -> A-layout), 4 WMMAs P*V.
// ---------------------------------------------------------------------------
__global__ __launch_bounds__(128)
void flash_attn_kernel(const _Float16* __restrict__ qh,
                       const _Float16* __restrict__ kh,
                       const _Float16* __restrict__ vh,
                       _Float16* __restrict__ ao) {
  constexpr int S = 2048, HD = 1024, BS = 2 * S;
  const int b    = blockIdx.z;
  const int h    = blockIdx.y;
  const int tid  = threadIdx.x;
  const int wave = tid >> 5;
  const int lane = tid & 31;
  const int hi   = lane >> 4;
  const int l16  = lane & 15;
  const int kbA  = hi * 8;
  const int kbB  = hi * 16;
  const int s0   = blockIdx.x * 64 + wave * 16;

  __shared__ __align__(16) _Float16 ldsV[32 * 64];
  __shared__ __align__(16) _Float16 ldsP[4][16 * 32];

  // Q fragments, fixed for whole key loop (row = s0 + l16, feature dim 64 = 2x32)
  v16h qA[2];
  {
    const _Float16* qb = qh + (size_t)(b * S + s0 + l16) * HD + h * 64;
#pragma unroll
    for (int ks = 0; ks < 2; ++ks) {
      v8h lo  = *(const v8h*)(qb + ks * 32 + kbB);
      v8h hi8 = *(const v8h*)(qb + ks * 32 + kbB + 8);
      qA[ks] = __builtin_shufflevector(lo, hi8, SHUF16);
    }
  }

  // Tensor DMA descriptor group 1 (loop-invariant):
  //   data_size=2B, tensor_dim0=HD, tensor_dim1=B*S, tile=64x32,
  //   tensor_dim0_stride=HD, no padding/iteration/multicast.
  u32x8 g1;
  g1[0] = 0x00010000u;                                   // data_size code 1 (2B)
  g1[1] = ((unsigned)HD & 0xFFFFu) << 16;                // tensor_dim0[15:0]
  g1[2] = ((unsigned)HD >> 16) | (((unsigned)BS & 0xFFFFu) << 16);  // dim0 hi | dim1 lo
  g1[3] = ((unsigned)BS >> 16) | (64u << 16);            // dim1 hi | tile_dim0=64
  g1[4] = 32u;                                           // tile_dim1=32, tile_dim2=0
  g1[5] = (unsigned)HD;                                  // tensor_dim0_stride lo32
  g1[6] = 0u;
  g1[7] = 0u;
  const unsigned ldsVaddr = (unsigned)(size_t)(void*)&ldsV[0];

  v8f acc[4] = {};
  float mrow[8], lrow[8];
#pragma unroll
  for (int r = 0; r < 8; ++r) { mrow[r] = -__builtin_inff(); lrow[r] = 0.0f; }

  for (int t0 = 0; t0 < S; t0 += 32) {
    __syncthreads();  // previous iteration's ldsV readers done
    if (wave == 0) {  // TDM: stage V tile [32 keys x 64 features] into LDS
      const unsigned long long ga =
          (unsigned long long)(size_t)(vh + (size_t)(b * S + t0) * HD + h * 64);
      u32x4 g0;
      g0[0] = 1u;                                         // count=1, user mode
      g0[1] = ldsVaddr;                                   // LDS byte address
      g0[2] = (unsigned)ga;                               // global_addr[31:0]
      g0[3] = (unsigned)((ga >> 32) & 0x01FFFFFFu) | (2u << 30);  // addr hi | type=2
      asm volatile("tensor_load_to_lds %0, %1" :: "s"(g0), "s"(g1) : "memory");
      __builtin_amdgcn_s_wait_tensorcnt(0);
    }
    __syncthreads();  // V tile visible to all waves

    // Scores: two 16x16 C-tiles covering key cols [t0, t0+32)
    v8f c1 = {}, c2 = {};
    {
      const _Float16* kb0 = kh + (size_t)(b * S + t0 + l16) * HD + h * 64 + kbB;
      const _Float16* kb1 = kh + (size_t)(b * S + t0 + 16 + l16) * HD + h * 64 + kbB;
      v16h bk00 = *(const v16h*)(kb0);
      v16h bk10 = *(const v16h*)(kb1);
      v16h bk01 = *(const v16h*)(kb0 + 32);
      v16h bk11 = *(const v16h*)(kb1 + 32);
      if (t0 + 32 < S) { __builtin_prefetch(kb0 + 32 * HD, 0, 3); }
      c1 = wmma_f16(qA[0], bk00, c1);
      c2 = wmma_f16(qA[0], bk10, c2);
      c1 = wmma_f16(qA[1], bk01, c1);
      c2 = wmma_f16(qA[1], bk11, c2);
    }

    // Online softmax: row r's 32 columns live in c1[r], c2[r] across the 16
    // lanes of this lane-half; DPP butterflies reduce within the half.
    constexpr float scale = 0.125f;  // 1/sqrt(64)
    float p1[8], p2[8], sc[8];
#pragma unroll
    for (int r = 0; r < 8; ++r) {
      const float s1 = c1[r] * scale, s2 = c2[r] * scale;
      const float tmax = red_max16(fmaxf(s1, s2));
      const float mnew = fmaxf(mrow[r], tmax);
      const float scl  = __expf(mrow[r] - mnew);
      const float e1   = __expf(s1 - mnew);
      const float e2   = __expf(s2 - mnew);
      const float rs   = red_sum16(e1 + e2);
      lrow[r] = lrow[r] * scl + rs;
      mrow[r] = mnew;
      p1[r] = e1; p2[r] = e2; sc[r] = scl;
    }
#pragma unroll
    for (int et = 0; et < 4; ++et)
#pragma unroll
      for (int r = 0; r < 8; ++r) acc[et][r] *= sc[r];

    // P: C-layout -> A-layout via per-wave LDS (DS ops in-order per wave)
#pragma unroll
    for (int r = 0; r < 8; ++r) {
      const int row = r + hi * 8;
      ldsP[wave][row * 32 + l16]      = (_Float16)p1[r];
      ldsP[wave][row * 32 + 16 + l16] = (_Float16)p2[r];
    }
    v16h aP;
    {
      const _Float16* pb = &ldsP[wave][l16 * 32 + kbB];
      v8h lo  = *(const v8h*)(pb);
      v8h hi8 = *(const v8h*)(pb + 8);
      aP = __builtin_shufflevector(lo, hi8, SHUF16);
    }

    // O += P(16x32) * V(32x64): 4 WMMAs, B frags gathered from ldsV
#pragma unroll
    for (int et = 0; et < 4; ++et) {
      v16h bv;
#pragma unroll
      for (int j = 0; j < 16; ++j)
        bv[j] = ldsV[(kbB + j) * 64 + et * 16 + l16];
      acc[et] = wmma_f16(aP, bv, acc[et]);
    }
  }

  // Normalize and write out f16 [B*S, H*64]
  float inv[8];
#pragma unroll
  for (int r = 0; r < 8; ++r) inv[r] = 1.0f / lrow[r];
#pragma unroll
  for (int et = 0; et < 4; ++et) {
#pragma unroll
    for (int r = 0; r < 8; ++r) {
      const int row = s0 + r + hi * 8;
      const int col = h * 64 + et * 16 + l16;
      ao[(size_t)(b * S + row) * HD + col] = (_Float16)(acc[et][r] * inv[r]);
    }
  }
}

// ---------------------------------------------------------------------------
extern "C" void kernel_launch(void* const* d_in, const int* in_sizes, int n_in,
                              void* d_out, int out_size, void* d_ws, size_t ws_size,
                              hipStream_t stream) {
  (void)in_sizes; (void)n_in; (void)out_size; (void)ws_size;
  const float* query = (const float*)d_in[0];
  const float* key   = (const float*)d_in[1];
  const float* value = (const float*)d_in[2];
  const float* Wq    = (const float*)d_in[3];
  const float* bq    = (const float*)d_in[4];
  const float* Wk    = (const float*)d_in[5];
  const float* bk    = (const float*)d_in[6];
  const float* Wv    = (const float*)d_in[7];
  const float* bv    = (const float*)d_in[8];
  const float* Wo    = (const float*)d_in[9];
  const float* bo    = (const float*)d_in[10];

  constexpr size_t ELEMS = (size_t)4096 * 1024;  // B*S x H*64
  _Float16* qh  = (_Float16*)d_ws;
  _Float16* khp = qh + ELEMS;
  _Float16* vhp = khp + ELEMS;
  _Float16* aop = vhp + ELEMS;   // total 32 MB of d_ws

  const dim3 gGemm(1024 / 64, 4096 / 64);  // (N tiles, M tiles)
  const dim3 blk(128);

  // Q/K/V projections (f32 A, per-head weight layout, f16 out, fused bias)
  gemm_kernel<true, true, false><<<gGemm, blk, 0, stream>>>(query, Wq, bq, qh);
  gemm_kernel<true, true, false><<<gGemm, blk, 0, stream>>>(key,   Wk, bk, khp);
  gemm_kernel<true, true, false><<<gGemm, blk, 0, stream>>>(value, Wv, bv, vhp);

  // Attention (TDM-staged V tiles)
  flash_attn_kernel<<<dim3(2048 / 64, 16, 2), blk, 0, stream>>>(qh, khp, vhp, aop);

  // Output projection (f16 A, row-major Wo, f32 out + bo) -> d_out
  gemm_kernel<false, false, true><<<gGemm, blk, 0, stream>>>(aop, Wo, bo, (float*)d_out);
}
